// GroupedQueryAttention_49168785605457
// MI455X (gfx1250) — matrix operand order repaired and checked by reading, not compiled, not run on hardware
//
#include <hip/hip_runtime.h>
#include <hip/hip_bf16.h>
#include <stdint.h>

// Problem dims (match reference)
#define Hdim 2048
#define Gh   16
#define Dh   128     // head dim
#define Bb   2
#define Tt   2048
#define Mrows (Bb*Tt)   // 4096 flattened rows
#define LNEPS 1e-5f

typedef unsigned short u16;
typedef __attribute__((ext_vector_type(16))) __bf16 v16bf;
typedef __attribute__((ext_vector_type(8)))  float  v8f;

union FragU { uint4 q[2]; v16bf v; };

__device__ __forceinline__ u16 f2bf(float x) {
  unsigned u = __float_as_uint(x);
  unsigned r = u + 0x7fffu + ((u >> 16) & 1u);  // round-to-nearest-even
  return (u16)(r >> 16);
}

// ---- CDNA5 async global->LDS copy (ASYNCcnt path), 16B per lane -----------
// LDS destination address = low 32 bits of the flat shared pointer (ISA 10.2:
// flat->LDS mapping truncates to addr[31:0]).
__device__ __forceinline__ unsigned lds_addr(const void* p) {
  return (unsigned)(uintptr_t)p;
}
__device__ __forceinline__ void async_ld16(unsigned lds, const void* g) {
  asm volatile("global_load_async_to_lds_b128 %0, %1, off"
               :: "v"(lds), "v"((unsigned long long)(uintptr_t)g)
               : "memory");
}
__device__ __forceinline__ void wait_async0() {
  asm volatile("s_wait_asynccnt 0x0" ::: "memory");
}

// A-matrix fragment (16x32 bf16). Per CDNA5 ISA 7.12.2:
// lane l (row = l&15, half = l>>4): VGPR0..3 hold K = half*8 .. half*8+7,
// VGPR4..7 hold K = 16+half*8 .. 16+half*8+7  -> two 16B chunks.
__device__ __forceinline__ v16bf ld_a_frag(const u16* rowbase, int kc, int half) {
  const char* p = (const char*)rowbase;
  int o = kc * 64 + half * 16;
  FragU f;
  f.q[0] = *(const uint4*)(p + o);
  f.q[1] = *(const uint4*)(p + o + 32);
  return f.v;
}

// B-matrix fragment (32x16 bf16), operand stored as Bt[n][k] (K contiguous):
// lane l (col = l&15, half = l>>4) holds K = half*8 .. half*8+7, 16+half*8 .. 16+half*8+7 -> two 16B chunks.
__device__ __forceinline__ v16bf ld_b_frag(const u16* rowbase, int kc, int half) {
  const char* p = (const char*)rowbase;
  int o = kc * 64 + half * 16;
  FragU f;
  f.q[0] = *(const uint4*)(p + o);
  f.q[1] = *(const uint4*)(p + o + 32);
  return f.v;
}

__device__ __forceinline__ v8f wmma_bf16(v16bf a, v16bf b, v8f c) {
  return __builtin_amdgcn_wmma_f32_16x16x32_bf16(false, a, false, b, (short)0, c, false, false);
}

// ---------------------------------------------------------------------------
// Elementwise f32 -> bf16 convert (for x)
// ---------------------------------------------------------------------------
__global__ __launch_bounds__(256) void cvt_bf16_kernel(const float* __restrict__ in,
                                                       u16* __restrict__ out, int n) {
  int i = (blockIdx.x * 256 + threadIdx.x) * 4;
  if (i < n) {
    float4 v = *(const float4*)&in[i];
    ushort4 o;
    o.x = f2bf(v.x); o.y = f2bf(v.y); o.z = f2bf(v.z); o.w = f2bf(v.w);
    *(ushort4*)&out[i] = o;
  }
}

// ---------------------------------------------------------------------------
// Transpose + convert: W[k][n] f32 (2048x2048) -> Wt[n][k] bf16
// ---------------------------------------------------------------------------
__global__ __launch_bounds__(256) void transpose_cvt_kernel(const float* __restrict__ W,
                                                            u16* __restrict__ Wt) {
  __shared__ float tile[64][65];
  const int tid = threadIdx.x;
  const int kb = blockIdx.x * 64, nb = blockIdx.y * 64;
#pragma unroll
  for (int i = 0; i < 16; i++) {
    int idx = tid + i * 256;
    int r = idx >> 6, c = idx & 63;
    tile[r][c] = W[(size_t)(kb + r) * Hdim + nb + c];
  }
  __syncthreads();
#pragma unroll
  for (int i = 0; i < 16; i++) {
    int idx = tid + i * 256;
    int r = idx >> 6, c = idx & 63;   // r = n-local, c = k-local
    Wt[(size_t)(nb + r) * Hdim + kb + c] = f2bf(tile[c][r]);
  }
}

// ---------------------------------------------------------------------------
// Tiled WMMA GEMM, 128x128 output tile. 8 waves in a 4(M) x 2(N) grid; each
// wave computes 32x64 (2 A-frags x 4 B-frags -> 8 WMMAs per 32-K chunk with
// only 12 ds_load_b128). Double-buffered LDS tiles filled by async DMA.
// A:[M][K] bf16 row-major, Bt:[N][K] bf16 (B transposed).
// mode 0: out Q/K head-major  [bg][t][d]  (bf16)
// mode 1: out V  transposed   [bg][d][t]  (bf16)
// mode 2: out-proj f32 epilogue: out = acc + bias[n] + resid[m][n]
// ---------------------------------------------------------------------------
__global__ __launch_bounds__(256)
void gemm_bf16_128x128(const u16* __restrict__ A, const u16* __restrict__ Bt,
                       int K, int mode,
                       u16* __restrict__ outb, float* __restrict__ outf,
                       const float* __restrict__ resid, const float* __restrict__ bias) {
  __shared__ __align__(16) u16 sA[2][128][72];   // 64 K-block + 8 pad (144B rows)
  __shared__ __align__(16) u16 sB[2][128][72];

  const int tid = threadIdx.x;
  const int w = tid >> 5, lane = tid & 31;
  const int half = lane >> 4, l16 = lane & 15;
  const int wm = w >> 1, wn = w & 1;             // wave grid 4 x 2
  const int m0 = blockIdx.x * 128, n0 = blockIdx.y * 128;

  // This thread's fixed slice of the cooperative tile loads (4 x 16B each op)
  int ldRow[4], ldCol[4];
#pragma unroll
  for (int i = 0; i < 4; i++) {
    int seg = tid + i * 256;               // 1024 16B segments per operand
    ldRow[i] = seg >> 3;
    ldCol[i] = (seg & 7) * 8;
  }

  v8f acc[2][4];
#pragma unroll
  for (int mi = 0; mi < 2; mi++)
#pragma unroll
    for (int f = 0; f < 4; f++)
#pragma unroll
      for (int r = 0; r < 8; r++) acc[mi][f][r] = 0.f;

  // Prologue: async-fill buffer 0 with the first K-slab
#pragma unroll
  for (int i = 0; i < 4; i++) {
    async_ld16(lds_addr(&sA[0][ldRow[i]][ldCol[i]]), &A[(size_t)(m0 + ldRow[i]) * K + ldCol[i]]);
    async_ld16(lds_addr(&sB[0][ldRow[i]][ldCol[i]]), &Bt[(size_t)(n0 + ldRow[i]) * K + ldCol[i]]);
  }

  int cur = 0;
  for (int kb = 0; kb < K; kb += 64) {
    wait_async0();
    __syncthreads();                       // tile `cur` visible to all waves
    if (kb + 64 < K) {                     // kick off next slab into other buffer
      int nxt = cur ^ 1, kn = kb + 64;
#pragma unroll
      for (int i = 0; i < 4; i++) {
        async_ld16(lds_addr(&sA[nxt][ldRow[i]][ldCol[i]]), &A[(size_t)(m0 + ldRow[i]) * K + kn + ldCol[i]]);
        async_ld16(lds_addr(&sB[nxt][ldRow[i]][ldCol[i]]), &Bt[(size_t)(n0 + ldRow[i]) * K + kn + ldCol[i]]);
      }
    }
#pragma unroll
    for (int kc = 0; kc < 2; kc++) {
      v16bf a0 = ld_a_frag(&sA[cur][wm * 32 + l16][0], kc, half);
      v16bf a1 = ld_a_frag(&sA[cur][wm * 32 + 16 + l16][0], kc, half);
      v16bf bfr[4];
#pragma unroll
      for (int f = 0; f < 4; f++)
        bfr[f] = ld_b_frag(&sB[cur][wn * 64 + f * 16 + l16][0], kc, half);
      __builtin_amdgcn_sched_barrier(0);   // batch the ds_loads; one dscnt wait
#pragma unroll
      for (int f = 0; f < 4; f++) acc[0][f] = wmma_bf16(a0, bfr[f], acc[0][f]);
#pragma unroll
      for (int f = 0; f < 4; f++) acc[1][f] = wmma_bf16(a1, bfr[f], acc[1][f]);
    }
    cur ^= 1;
  }

  // C/D layout: lane = column; VGPR r = row (m = ... + half*8 + r)
#pragma unroll
  for (int mi = 0; mi < 2; mi++) {
#pragma unroll
    for (int f = 0; f < 4; f++) {
      int n = n0 + wn * 64 + f * 16 + l16;
#pragma unroll
      for (int r = 0; r < 8; r++) {
        int m = m0 + wm * 32 + mi * 16 + half * 8 + r;
        float v = acc[mi][f][r];
        if (mode == 2) {
          outf[(size_t)m * Hdim + n] = v + bias[n] + resid[(size_t)m * Hdim + n];
        } else {
          int b = m >> 11, t = m & (Tt - 1);
          int g = n >> 7, d = n & (Dh - 1);
          int bg = b * Gh + g;
          size_t idx = (mode == 0) ? ((size_t)bg * Tt + t) * Dh + d
                                   : ((size_t)bg * Dh + d) * Tt + t;
          outb[idx] = f2bf(v);
        }
      }
    }
  }
}

// ---------------------------------------------------------------------------
// Flash attention: one block = 128 query rows of one (b,g) head.
// Q tile resident; K/V streamed in double-buffered async s-tiles of 64.
// Each wave owns 16 complete rows -> row stats via width-16 shuffle reductions.
// ---------------------------------------------------------------------------
__global__ __launch_bounds__(256)
void attn_kernel(const u16* __restrict__ Qb, const u16* __restrict__ Kb,
                 const u16* __restrict__ Vtb, u16* __restrict__ Ob) {
  __shared__ __align__(16) u16 sQ[128][136];     // 128 d + 8 pad (272B rows)
  __shared__ __align__(16) u16 sK[2][64][136];
  __shared__ __align__(16) u16 sV[2][128][72];   // Vt: rows = d, cols = s (64 + pad)
  __shared__ __align__(16) u16 sP[128][72];      // probs, row-major [m][s]

  const int tid = threadIdx.x;
  const int w = tid >> 5, lane = tid & 31;
  const int half = lane >> 4, l16 = lane & 15;
  const int bg = blockIdx.y;
  const int q0 = blockIdx.x * 128;
  const size_t baseQK = (size_t)bg * Tt * Dh;
  const size_t baseV  = (size_t)bg * Dh * Tt;

  // Per-thread slices of the cooperative K/V tile loads
  int rk[4], ck[4], rv[4], cv[4];
#pragma unroll
  for (int i = 0; i < 4; i++) {
    int seg = tid + i * 256;
    rk[i] = seg >> 4; ck[i] = (seg & 15) * 8;   // K: 64 rows x 16 segs
    rv[i] = seg >> 3; cv[i] = (seg & 7) * 8;    // Vt: 128 rows x 8 segs
  }

  // Prologue: async-load Q tile (128x128) and s-tile 0 of K/V into buffer 0
#pragma unroll
  for (int i = 0; i < 8; i++) {
    int seg = tid + i * 256;                    // 2048 16B segments
    int row = seg >> 4, col = (seg & 15) * 8;
    async_ld16(lds_addr(&sQ[row][col]), &Qb[baseQK + (size_t)(q0 + row) * Dh + col]);
  }
#pragma unroll
  for (int i = 0; i < 4; i++) {
    async_ld16(lds_addr(&sK[0][rk[i]][ck[i]]), &Kb[baseQK + (size_t)rk[i] * Dh + ck[i]]);
    async_ld16(lds_addr(&sV[0][rv[i]][cv[i]]), &Vtb[baseV + (size_t)rv[i] * Tt + cv[i]]);
  }

  float mrow[8], lrow[8];
  v8f oacc[8];
#pragma unroll
  for (int r = 0; r < 8; r++) { mrow[r] = -1e30f; lrow[r] = 0.f; }
#pragma unroll
  for (int f = 0; f < 8; f++)
#pragma unroll
    for (int r = 0; r < 8; r++) oacc[f][r] = 0.f;

  const float scale = 0.08838834764831845f;  // D^-0.5, D=128

  int cur = 0;
  for (int s0 = 0; s0 < Tt; s0 += 64) {
    wait_async0();
    __syncthreads();                           // tile `cur` (+ Q on iter 0) ready
    if (s0 + 64 < Tt) {                        // prefetch next s-tile into other buffer
      int nxt = cur ^ 1, sn = s0 + 64;
#pragma unroll
      for (int i = 0; i < 4; i++) {
        async_ld16(lds_addr(&sK[nxt][rk[i]][ck[i]]), &Kb[baseQK + (size_t)(sn + rk[i]) * Dh + ck[i]]);
        async_ld16(lds_addr(&sV[nxt][rv[i]][cv[i]]), &Vtb[baseV + (size_t)rv[i] * Tt + sn + cv[i]]);
      }
    }

    // S = Q * K^T  (per wave: 16 x 64, K-dim = Dh = 128 -> 4 chunks of 32)
    v8f sacc[4];
#pragma unroll
    for (int f = 0; f < 4; f++)
#pragma unroll
      for (int r = 0; r < 8; r++) sacc[f][r] = 0.f;
#pragma unroll
    for (int kc = 0; kc < 4; kc++) {
      v16bf a = ld_a_frag(&sQ[w * 16 + l16][0], kc, half);
      v16bf bfr[4];
#pragma unroll
      for (int f = 0; f < 4; f++) bfr[f] = ld_b_frag(&sK[cur][f * 16 + l16][0], kc, half);
      __builtin_amdgcn_sched_barrier(0);
#pragma unroll
      for (int f = 0; f < 4; f++) sacc[f] = wmma_bf16(a, bfr[f], sacc[f]);
    }

    // Online softmax. Lane's 8 rows are (half*8 + r) of this wave's strip.
#pragma unroll
    for (int f = 0; f < 4; f++)
#pragma unroll
      for (int r = 0; r < 8; r++) sacc[f][r] *= scale;

#pragma unroll
    for (int r = 0; r < 8; r++) {
      float tm = fmaxf(fmaxf(sacc[0][r], sacc[1][r]), fmaxf(sacc[2][r], sacc[3][r]));
#pragma unroll
      for (int off = 8; off; off >>= 1) tm = fmaxf(tm, __shfl_xor(tm, off, 16));
      float mnew = fmaxf(mrow[r], tm);
      float fac = __expf(mrow[r] - mnew);
      float ts = 0.f;
#pragma unroll
      for (int f = 0; f < 4; f++) {
        float p = __expf(sacc[f][r] - mnew);
        sacc[f][r] = p;
        ts += p;
      }
#pragma unroll
      for (int off = 8; off; off >>= 1) ts += __shfl_xor(ts, off, 16);
      lrow[r] = lrow[r] * fac + ts;
      mrow[r] = mnew;
#pragma unroll
      for (int f = 0; f < 8; f++) oacc[f][r] *= fac;
    }

    // Spill P (bf16) to this wave's own 16 LDS rows (no cross-wave hazard).
#pragma unroll
    for (int f = 0; f < 4; f++)
#pragma unroll
      for (int r = 0; r < 8; r++)
        sP[w * 16 + half * 8 + r][f * 16 + l16] = f2bf(sacc[f][r]);

    // O += P * V   (K-dim = 64 -> 2 chunks; N = Dh = 128 -> 8 frags)
#pragma unroll
    for (int kc = 0; kc < 2; kc++) {
      v16bf a = ld_a_frag(&sP[w * 16 + l16][0], kc, half);
      v16bf bfr[8];
#pragma unroll
      for (int f = 0; f < 8; f++) bfr[f] = ld_b_frag(&sV[cur][f * 16 + l16][0], kc, half);
      __builtin_amdgcn_sched_barrier(0);
#pragma unroll
      for (int f = 0; f < 8; f++) oacc[f] = wmma_bf16(a, bfr[f], oacc[f]);
    }
    cur ^= 1;
  }

  // Epilogue: O /= l, scatter into [B,T,H] bf16 for the out-proj GEMM.
  const int b = bg / Gh, g = bg % Gh;
#pragma unroll
  for (int r = 0; r < 8; r++) {
    float inv = 1.f / lrow[r];
    int t = q0 + w * 16 + half * 8 + r;
    size_t rowbase = ((size_t)b * Tt + t) * Hdim + (size_t)g * Dh;
#pragma unroll
    for (int f = 0; f < 8; f++)
      Ob[rowbase + f * 16 + l16] = f2bf(oacc[f][r] * inv);
  }
}

// ---------------------------------------------------------------------------
// In-place LayerNorm over rows of 2048 f32
// ---------------------------------------------------------------------------
__global__ __launch_bounds__(256)
void ln_kernel(float* __restrict__ y, const float* __restrict__ gamma,
               const float* __restrict__ beta) {
  __shared__ float rs[256], rss[256];
  const int row = blockIdx.x, tid = threadIdx.x;
  const size_t base = (size_t)row * Hdim;
  float v[8], s = 0.f, ss = 0.f;
#pragma unroll
  for (int i = 0; i < 8; i++) {
    v[i] = y[base + tid + i * 256];
    s += v[i]; ss += v[i] * v[i];
  }
  rs[tid] = s; rss[tid] = ss;
  __syncthreads();
  for (int off = 128; off; off >>= 1) {
    if (tid < off) { rs[tid] += rs[tid + off]; rss[tid] += rss[tid + off]; }
    __syncthreads();
  }
  float mu = rs[0] * (1.f / Hdim);
  float var = rss[0] * (1.f / Hdim) - mu * mu;
  float inv = rsqrtf(var + LNEPS);
#pragma unroll
  for (int i = 0; i < 8; i++) {
    int c = tid + i * 256;
    y[base + c] = (v[i] - mu) * inv * gamma[c] + beta[c];
  }
}

// ---------------------------------------------------------------------------
extern "C" void kernel_launch(void* const* d_in, const int* in_sizes, int n_in,
                              void* d_out, int out_size, void* d_ws, size_t ws_size,
                              hipStream_t stream) {
  const float* x     = (const float*)d_in[0];
  const float* Wq    = (const float*)d_in[1];
  const float* Wk    = (const float*)d_in[2];
  const float* Wv    = (const float*)d_in[3];
  const float* Wp    = (const float*)d_in[4];
  const float* bp    = (const float*)d_in[5];
  const float* gamma = (const float*)d_in[6];
  const float* beta  = (const float*)d_in[7];
  float* out = (float*)d_out;

  char* ws = (char*)d_ws;
  size_t off = 0;
  auto alloc = [&](size_t bytes) {
    char* p = ws + off;
    off += (bytes + 255) & ~(size_t)255;
    return p;
  };
  u16* xb  = (u16*)alloc((size_t)Mrows * Hdim * 2);   // x bf16
  u16* Wqt = (u16*)alloc((size_t)Hdim * Hdim * 2);    // W^T bf16
  u16* Wkt = (u16*)alloc((size_t)Hdim * Hdim * 2);
  u16* Wvt = (u16*)alloc((size_t)Hdim * Hdim * 2);
  u16* Wpt = (u16*)alloc((size_t)Hdim * Hdim * 2);
  u16* Qb  = (u16*)alloc((size_t)Bb * Gh * Tt * Dh * 2);  // [bg][t][d]
  u16* Kb  = (u16*)alloc((size_t)Bb * Gh * Tt * Dh * 2);  // [bg][t][d]
  u16* Vtb = (u16*)alloc((size_t)Bb * Gh * Dh * Tt * 2);  // [bg][d][t]
  u16* Ob  = (u16*)alloc((size_t)Mrows * Hdim * 2);       // attention out [B,T,H]

  // 1) precision conversion / weight transposes
  cvt_bf16_kernel<<<(Mrows * Hdim) / 1024, 256, 0, stream>>>(x, xb, Mrows * Hdim);
  dim3 tg(Hdim / 64, Hdim / 64);
  transpose_cvt_kernel<<<tg, 256, 0, stream>>>(Wq, Wqt);
  transpose_cvt_kernel<<<tg, 256, 0, stream>>>(Wk, Wkt);
  transpose_cvt_kernel<<<tg, 256, 0, stream>>>(Wv, Wvt);
  transpose_cvt_kernel<<<tg, 256, 0, stream>>>(Wp, Wpt);

  // 2) QKV projections (WMMA GEMMs, async double-buffered)
  dim3 gg(Mrows / 128, Hdim / 128);
  gemm_bf16_128x128<<<gg, 256, 0, stream>>>(xb, Wqt, Hdim, 0, Qb, nullptr, nullptr, nullptr);
  gemm_bf16_128x128<<<gg, 256, 0, stream>>>(xb, Wkt, Hdim, 0, Kb, nullptr, nullptr, nullptr);
  gemm_bf16_128x128<<<gg, 256, 0, stream>>>(xb, Wvt, Hdim, 1, Vtb, nullptr, nullptr, nullptr);

  // 3) flash attention per (b,g) head
  dim3 ga(Tt / 128, Bb * Gh);
  attn_kernel<<<ga, 256, 0, stream>>>(Qb, Kb, Vtb, Ob);

  // 4) out-proj + bias + residual (f32 epilogue into d_out)
  gemm_bf16_128x128<<<gg, 256, 0, stream>>>(Ob, Wpt, Hdim, 2, nullptr, out, x, bp);

  // 5) LayerNorm in place
  ln_kernel<<<Mrows, 256, 0, stream>>>(out, gamma, beta);
}
